// Grouped_GEMM_MoE_41832981463195
// MI455X (gfx1250) — matrix operand order repaired and checked by reading, not compiled, not run on hardware
//
#include <hip/hip_runtime.h>

// ---------------------------------------------------------------------------
// MoE grouped GEMM for MI455X (gfx1250, wave32, WMMA).
//   T=16384 tokens, H=1024, F=2048, E=8, TOPK=2 -> 32768 routed rows.
// Pipeline:
//   convert x/w1/w2 -> f16 (weights in WMMA-B tiled layout, one-time pass)
//   deterministic counting sort of assignments
//   GEMM1 + exact GELU (v_wmma_f32_16x16x32_f16, f32 accum) -> f16 h
//   GEMM2 (+bias, +routing weight) -> per-assignment f32 rows
//   per-token combine (overwrites d_out, no atomics anywhere)
// GEMM main loops: ping-pong LDS staging (1 barrier/K-step) + unroll-2 to
// eliminate B-fragment register rotation.
// ---------------------------------------------------------------------------

typedef __attribute__((ext_vector_type(16))) _Float16 v16h;
typedef __attribute__((ext_vector_type(4)))  _Float16 v4h;
typedef __attribute__((ext_vector_type(8)))  float    v8f;

static constexpr int TT      = 16384;   // tokens
static constexpr int HH      = 1024;    // hidden
static constexpr int FF      = 2048;    // expert intermediate
static constexpr int EE      = 8;       // experts
static constexpr int TOPK    = 2;
static constexpr int AA      = TT * TOPK;    // 32768 routed assignments
static constexpr int BM      = 128;     // M tile (routed rows)
static constexpr int BN      = 128;     // N tile
static constexpr int KC      = 32;      // K per WMMA step
static constexpr int MAXTIL  = AA / BM + EE;   // 264 worst-case tiles
static constexpr int NCHUNK  = 64;      // routing chunks
static constexpr int CHSZ    = AA / NCHUNK;    // 512 assignments/chunk
static constexpr int XSTRIDE = 40;      // LDS halves per staged row (bank pad)

// ---------------- workspace layout (bytes) ----------------
static constexpr size_t WS_META  = 0;          // int meta[32]
static constexpr size_t WS_PART  = 256;        // int partial[EE*NCHUNK]
static constexpr size_t WS_COFF  = 4096;       // int chunk_off[EE*NCHUNK]
static constexpr size_t WS_TILES = 8192;       // int2 tiles[MAXTIL]
static constexpr size_t WS_SORT  = 16384;      // int sorted[AA]
static constexpr size_t WS_XH    = 262144;                                  // f16 x        (32 MB)
static constexpr size_t WS_W1C   = WS_XH  + (size_t)TT * HH * 2;            // f16 w1 tiled (32 MB)
static constexpr size_t WS_W2C   = WS_W1C + (size_t)EE * HH * FF * 2;       // f16 w2 tiled (32 MB)
static constexpr size_t WS_HBUF  = WS_W2C + (size_t)EE * FF * HH * 2;       // f16 h      (128 MB)
static constexpr size_t WS_OUT   = WS_HBUF + (size_t)AA * FF * 2;           // f32 rows   (128 MB)

// ---------------------------------------------------------------------------
// One-time conversions.
// ---------------------------------------------------------------------------
__global__ __launch_bounds__(256) void moe_convert_x(const float* __restrict__ src,
                                                     _Float16* __restrict__ dst) {
  const size_t total = (size_t)TT * HH / 4;
  for (size_t i = (size_t)blockIdx.x * blockDim.x + threadIdx.x; i < total;
       i += (size_t)gridDim.x * blockDim.x) {
    const float4 f = ((const float4*)src)[i];
    v4h h;
    h[0] = (_Float16)f.x; h[1] = (_Float16)f.y;
    h[2] = (_Float16)f.z; h[3] = (_Float16)f.w;
    ((v4h*)dst)[i] = h;
  }
}

// src: [E][K][N] f32 row-major.
// dst: [E][N/16][K/32][16][32] f16 -- B-fragment order used here (k = 16*khalf + s): a 32x16 WMMA B
// tile is one contiguous 1KB block, lane's 16 K-values are 32B contiguous.
__global__ __launch_bounds__(256) void moe_convert_w(const float* __restrict__ src,
                                                     _Float16* __restrict__ dst,
                                                     int K, int N) {
  const int kt = K >> 5;
  const size_t total = (size_t)EE * K * N / 4;
  for (size_t i = (size_t)blockIdx.x * blockDim.x + threadIdx.x; i < total;
       i += (size_t)gridDim.x * blockDim.x) {
    const size_t s = i * 4;
    const int e = (int)(s / ((size_t)K * N));
    const size_t rem = s - (size_t)e * K * N;
    const int k = (int)(rem / N);
    const int n = (int)(rem - (size_t)k * N);
    const float4 f = *(const float4*)(src + s);
    const float vv[4] = {f.x, f.y, f.z, f.w};
    _Float16* de = dst + (size_t)e * K * N;
#pragma unroll
    for (int q = 0; q < 4; ++q) {
      const int nn = n + q;
      de[((size_t)(nn >> 4) * kt + (k >> 5)) * 512 + (nn & 15) * 32 + (k & 31)] =
          (_Float16)vv[q];
    }
  }
}

// ---------------------------------------------------------------------------
// Routing pass 1: per-chunk expert histograms (LDS atomics -> deterministic).
// ---------------------------------------------------------------------------
__global__ __launch_bounds__(256) void moe_route_count(const int* __restrict__ topexp,
                                                       int* __restrict__ partial) {
  __shared__ int cnt[EE];
  if (threadIdx.x < EE) cnt[threadIdx.x] = 0;
  __syncthreads();
  const int c = blockIdx.x;
  for (int i = threadIdx.x; i < CHSZ; i += 256)
    atomicAdd(&cnt[topexp[c * CHSZ + i] & (EE - 1)], 1);
  __syncthreads();
  if (threadIdx.x < EE) partial[threadIdx.x * NCHUNK + c] = cnt[threadIdx.x];
}

// ---------------------------------------------------------------------------
// Routing pass 2: tiny serial scan -> bases, chunk offsets, tile list.
// ---------------------------------------------------------------------------
__global__ void moe_route_scan(const int* __restrict__ partial,
                               int* __restrict__ meta,
                               int* __restrict__ chunk_off,
                               int2* __restrict__ tiles) {
  if (threadIdx.x != 0 || blockIdx.x != 0) return;
  int cnt[EE], base[EE], acc = 0;
  for (int e = 0; e < EE; ++e) {
    int s = 0;
    for (int c = 0; c < NCHUNK; ++c) s += partial[e * NCHUNK + c];
    cnt[e] = s;
  }
  for (int e = 0; e < EE; ++e) { base[e] = acc; acc += cnt[e]; }
  for (int e = 0; e < EE; ++e) {
    int run = base[e];
    for (int c = 0; c < NCHUNK; ++c) {
      chunk_off[e * NCHUNK + c] = run;
      run += partial[e * NCHUNK + c];
    }
  }
  int nt = 0;
  for (int e = 0; e < EE; ++e)
    for (int k = 0; k < cnt[e]; k += BM) tiles[nt++] = make_int2(e, base[e] + k);
  meta[0] = nt;
  for (int e = 0; e < EE; ++e) { meta[8 + e] = cnt[e]; meta[16 + e] = base[e]; }
}

// ---------------------------------------------------------------------------
// Routing pass 3: stable scatter via wave32 ballots + cross-wave LDS scan.
// ---------------------------------------------------------------------------
__global__ __launch_bounds__(256) void moe_route_scatter(const int* __restrict__ topexp,
                                                         const int* __restrict__ chunk_off,
                                                         int* __restrict__ sorted) {
  __shared__ int chunkbase[EE];
  __shared__ int wavecnt[8][EE];
  const int c = blockIdx.x;
  const int tid = threadIdx.x, wave = tid >> 5, lane = tid & 31;
  if (tid < EE) chunkbase[tid] = chunk_off[tid * NCHUNK + c];
  __syncthreads();
  for (int round = 0; round < CHSZ / 256; ++round) {
    const int a = c * CHSZ + round * 256 + tid;
    const int e = topexp[a] & (EE - 1);
    unsigned bal[EE];
#pragma unroll
    for (int ee = 0; ee < EE; ++ee) bal[ee] = __builtin_amdgcn_ballot_w32(e == ee);
    if (lane == 0) {
#pragma unroll
      for (int ee = 0; ee < EE; ++ee) wavecnt[wave][ee] = __popc(bal[ee]);
    }
    __syncthreads();
    int off = chunkbase[e];
    for (int w = 0; w < wave; ++w) off += wavecnt[w][e];
    off += __popc(bal[e] & ((1u << lane) - 1u));
    sorted[off] = a;
    __syncthreads();
    if (tid < EE) {
      int s = 0;
      for (int w = 0; w < 8; ++w) s += wavecnt[w][tid];
      chunkbase[tid] += s;
    }
    __syncthreads();
  }
}

// ---------------------------------------------------------------------------
// WMMA helpers.
// ---------------------------------------------------------------------------
__device__ __forceinline__ v8f wmma_f16(v16h a, v16h b, v8f c) {
  return __builtin_amdgcn_wmma_f32_16x16x32_f16(false, a, false, b, (short)0, c,
                                                false, false);
}

// A 16x32 f16: lane<16 row=lane K{0..15}; lane>=16 row=lane-16 K{16..31}
__device__ __forceinline__ v16h load_a_lds(const _Float16* __restrict__ xs,
                                           int row, int khalf) {
  const _Float16* p = xs + row * XSTRIDE;
  v16h a;
#pragma unroll
  for (int j = 0; j < 8; ++j) {
    a[j]     = p[khalf * 16 + j];
    a[8 + j] = p[khalf * 16 + 8 + j];
  }
  return a;
}

// B from pre-tiled f16: this lane's 16 K-values are 32B contiguous.
__device__ __forceinline__ v16h load_b_f16(const _Float16* __restrict__ p) {
  union { float4 f[2]; v16h v; } u;
  u.f[0] = ((const float4*)p)[0];
  u.f[1] = ((const float4*)p)[1];
  return u.v;
}

__device__ __forceinline__ float gelu_exact(float v) {
  return 0.5f * v * (1.0f + erff(v * 0.70710678118654752f));
}

// ---------------------------------------------------------------------------
// GEMM1 + GELU: h[r,:] = gelu(x[tok(r),:] @ w1[e] + b1[e]) (f16 out).
// 8 waves in a 2x4 grid; each wave owns 64x32 = 4x2 WMMA fragments.
// ---------------------------------------------------------------------------
__global__ __launch_bounds__(256) void moe_gemm1(const _Float16* __restrict__ xh,
                                                 const _Float16* __restrict__ w1c,
                                                 const float* __restrict__ b1,
                                                 const int* __restrict__ sorted,
                                                 const int2* __restrict__ tiles,
                                                 const int* __restrict__ meta,
                                                 _Float16* __restrict__ hbuf) {
  const int tile = blockIdx.y;
  if (tile >= meta[0]) return;

  __shared__ _Float16 xs[2][BM * XSTRIDE];
  __shared__ int tok_lds[BM];

  const int e         = tiles[tile].x;
  const int row_start = tiles[tile].y;
  const int seg_end   = meta[16 + e] + meta[8 + e];
  int rows = seg_end - row_start;
  if (rows > BM) rows = BM;

  const _Float16* w1e = w1c + (size_t)e * HH * FF;
  const float*    b1e = b1 + (size_t)e * FF;
  const int nb = blockIdx.x * BN;

  const int tid = threadIdx.x;
  const int wave = tid >> 5, lane = tid & 31;
  const int wave_m = wave >> 2, wave_n = wave & 3;
  const int khalf = lane >> 4, laneN = lane & 15;

  if (tid < BM) {
    int r = row_start + tid;
    if (r > AA - 1) r = AA - 1;
    tok_lds[tid] = sorted[r] >> 1;  // TOPK == 2
  }
  __syncthreads();

  // staging: thread copies 32B of one row per K-step
  const int srow = tid >> 1, shalf = tid & 1;
  const _Float16* xsrc = xh + (size_t)tok_lds[srow] * HH + shalf * 16;
  float4* sdst[2] = {(float4*)&xs[0][srow * XSTRIDE + shalf * 16],
                     (float4*)&xs[1][srow * XSTRIDE + shalf * 16]};

  // B pointers: fragment base for this lane; +512 halves per K-chunk
  const _Float16* bptr[2];
#pragma unroll
  for (int ni = 0; ni < 2; ++ni)
    bptr[ni] = w1e + (size_t)((nb + wave_n * 32 + ni * 16) >> 4) * (HH / 32) * 512 +
               laneN * 32 + khalf * 16;

  constexpr int NK = HH / KC;
  v8f acc[4][2] = {};

  // prologue: chunk 0 -> buf0, preload chunk 1 into registers
  float4 sr0 = ((const float4*)xsrc)[0];
  float4 sr1 = ((const float4*)xsrc)[1];
  sdst[0][0] = sr0;
  sdst[0][1] = sr1;
  {
    const float4* s4 = (const float4*)(xsrc + KC);
    sr0 = s4[0];
    sr1 = s4[1];
  }
  v16h bf0 = load_b_f16(bptr[0]);
  v16h bf1 = load_b_f16(bptr[1]);

#pragma unroll 2
  for (int kc = 0; kc < NK; ++kc) {
    __syncthreads();  // buf[kc&1] staged; prior reads of buf[(kc+1)&1] done
    if (kc + 1 < NK) {
      float4* d = sdst[(kc + 1) & 1];
      d[0] = sr0;
      d[1] = sr1;
      if (kc + 2 < NK) {
        const float4* s4 = (const float4*)(xsrc + (kc + 2) * KC);
        sr0 = s4[0];
        sr1 = s4[1];
      }
    }

    v16h nb0 = bf0, nb1 = bf1;
    if (kc + 1 < NK) {
      nb0 = load_b_f16(bptr[0] + (size_t)(kc + 1) * 512);
      nb1 = load_b_f16(bptr[1] + (size_t)(kc + 1) * 512);
    }

    const _Float16* ab = &xs[kc & 1][0];
    v16h af[4];
#pragma unroll
    for (int mi = 0; mi < 4; ++mi)
      af[mi] = load_a_lds(ab, wave_m * 64 + mi * 16 + laneN, khalf);

#pragma unroll
    for (int mi = 0; mi < 4; ++mi) {
      acc[mi][0] = wmma_f16(af[mi], bf0, acc[mi][0]);
      acc[mi][1] = wmma_f16(af[mi], bf1, acc[mi][1]);
    }
    bf0 = nb0;
    bf1 = nb1;
  }

  // epilogue: bias + exact GELU -> f16 h (C layout: VGPR v -> row v / v+8)
#pragma unroll
  for (int ni = 0; ni < 2; ++ni) {
    const int col = nb + wave_n * 32 + ni * 16 + laneN;
    const float bias = b1e[col];
#pragma unroll
    for (int mi = 0; mi < 4; ++mi) {
#pragma unroll
      for (int v = 0; v < 8; ++v) {
        const int lr = wave_m * 64 + mi * 16 + khalf * 8 + v;
        if (lr < rows) {
          hbuf[(size_t)(row_start + lr) * FF + col] =
              (_Float16)gelu_exact(acc[mi][ni][v] + bias);
        }
      }
    }
  }
}

// ---------------------------------------------------------------------------
// GEMM2: wsout[a,:] = fw[a] * (h[r,:] @ w2[e] + b2[e])   (f32 out, no atomics)
// ---------------------------------------------------------------------------
__global__ __launch_bounds__(256) void moe_gemm2(const _Float16* __restrict__ hbuf,
                                                 const _Float16* __restrict__ w2c,
                                                 const float* __restrict__ b2,
                                                 const int* __restrict__ sorted,
                                                 const int2* __restrict__ tiles,
                                                 const int* __restrict__ meta,
                                                 const float* __restrict__ expw,
                                                 float* __restrict__ wsout) {
  const int tile = blockIdx.y;
  if (tile >= meta[0]) return;

  __shared__ _Float16 hs[2][BM * XSTRIDE];
  __shared__ int a_lds[BM];
  __shared__ float fw_lds[BM];

  const int e         = tiles[tile].x;
  const int row_start = tiles[tile].y;
  const int seg_end   = meta[16 + e] + meta[8 + e];
  int rows = seg_end - row_start;
  if (rows > BM) rows = BM;

  const _Float16* w2e = w2c + (size_t)e * FF * HH;
  const float*    b2e = b2 + (size_t)e * HH;
  const int nb = blockIdx.x * BN;

  const int tid = threadIdx.x;
  const int wave = tid >> 5, lane = tid & 31;
  const int wave_m = wave >> 2, wave_n = wave & 3;
  const int khalf = lane >> 4, laneN = lane & 15;

  if (tid < BM) {
    int r = row_start + tid;
    if (r > AA - 1) r = AA - 1;
    const int a = sorted[r];
    a_lds[tid] = a;
    fw_lds[tid] = expw[a];
  }

  const int srow = tid >> 1, shalf = tid & 1;
  int rg = row_start + srow;
  if (rg > AA - 1) rg = AA - 1;
  const _Float16* hsrc = hbuf + (size_t)rg * FF + shalf * 16;
  float4* sdst[2] = {(float4*)&hs[0][srow * XSTRIDE + shalf * 16],
                     (float4*)&hs[1][srow * XSTRIDE + shalf * 16]};

  const _Float16* bptr[2];
#pragma unroll
  for (int ni = 0; ni < 2; ++ni)
    bptr[ni] = w2e + (size_t)((nb + wave_n * 32 + ni * 16) >> 4) * (FF / 32) * 512 +
               laneN * 32 + khalf * 16;

  constexpr int NK = FF / KC;
  v8f acc[4][2] = {};

  float4 sr0 = ((const float4*)hsrc)[0];
  float4 sr1 = ((const float4*)hsrc)[1];
  sdst[0][0] = sr0;
  sdst[0][1] = sr1;
  {
    const float4* s4 = (const float4*)(hsrc + KC);
    sr0 = s4[0];
    sr1 = s4[1];
  }
  v16h bf0 = load_b_f16(bptr[0]);
  v16h bf1 = load_b_f16(bptr[1]);

#pragma unroll 2
  for (int kc = 0; kc < NK; ++kc) {
    __syncthreads();
    if (kc + 1 < NK) {
      float4* d = sdst[(kc + 1) & 1];
      d[0] = sr0;
      d[1] = sr1;
      if (kc + 2 < NK) {
        const float4* s4 = (const float4*)(hsrc + (kc + 2) * KC);
        sr0 = s4[0];
        sr1 = s4[1];
      }
    }

    v16h nb0 = bf0, nb1 = bf1;
    if (kc + 1 < NK) {
      nb0 = load_b_f16(bptr[0] + (size_t)(kc + 1) * 512);
      nb1 = load_b_f16(bptr[1] + (size_t)(kc + 1) * 512);
    }

    const _Float16* ab = &hs[kc & 1][0];
    v16h af[4];
#pragma unroll
    for (int mi = 0; mi < 4; ++mi)
      af[mi] = load_a_lds(ab, wave_m * 64 + mi * 16 + laneN, khalf);

#pragma unroll
    for (int mi = 0; mi < 4; ++mi) {
      acc[mi][0] = wmma_f16(af[mi], bf0, acc[mi][0]);
      acc[mi][1] = wmma_f16(af[mi], bf1, acc[mi][1]);
    }
    bf0 = nb0;
    bf1 = nb1;
  }

#pragma unroll
  for (int ni = 0; ni < 2; ++ni) {
    const int col = nb + wave_n * 32 + ni * 16 + laneN;
    const float bias = b2e[col];
#pragma unroll
    for (int mi = 0; mi < 4; ++mi) {
#pragma unroll
      for (int v = 0; v < 8; ++v) {
        const int lr = wave_m * 64 + mi * 16 + khalf * 8 + v;
        if (lr < rows) {
          wsout[(size_t)a_lds[lr] * HH + col] =
              (acc[mi][ni][v] + bias) * fw_lds[lr];
        }
      }
    }
  }
}

// ---------------------------------------------------------------------------
// Combine: y[t] = wsout[2t] + wsout[2t+1]. Overwrites d_out, deterministic.
// ---------------------------------------------------------------------------
__global__ __launch_bounds__(256) void moe_combine(const float* __restrict__ wsout,
                                                   float* __restrict__ y) {
  const size_t t = blockIdx.x;
  const int ci = threadIdx.x;
  const float4 a = ((const float4*)(wsout + (2 * t) * HH))[ci];
  const float4 b = ((const float4*)(wsout + (2 * t + 1) * HH))[ci];
  float4 o;
  o.x = a.x + b.x; o.y = a.y + b.y; o.z = a.z + b.z; o.w = a.w + b.w;
  ((float4*)(y + t * HH))[ci] = o;
}

// ---------------------------------------------------------------------------
extern "C" void kernel_launch(void* const* d_in, const int* in_sizes, int n_in,
                              void* d_out, int out_size, void* d_ws, size_t ws_size,
                              hipStream_t stream) {
  const float* x    = (const float*)d_in[0];  // [T, H]
  const float* expw = (const float*)d_in[1];  // [T, TOPK]
  const int*   tope = (const int*)d_in[2];    // [T, TOPK]
  const float* w1   = (const float*)d_in[3];  // [E, H, F]
  const float* b1   = (const float*)d_in[4];  // [E, F]
  const float* w2   = (const float*)d_in[5];  // [E, F, H]
  const float* b2   = (const float*)d_in[6];  // [E, H]
  float* y = (float*)d_out;

  char* ws = (char*)d_ws;
  int*      meta      = (int*)(ws + WS_META);
  int*      partial   = (int*)(ws + WS_PART);
  int*      chunk_off = (int*)(ws + WS_COFF);
  int2*     tiles     = (int2*)(ws + WS_TILES);
  int*      sorted    = (int*)(ws + WS_SORT);
  _Float16* xh        = (_Float16*)(ws + WS_XH);
  _Float16* w1c       = (_Float16*)(ws + WS_W1C);
  _Float16* w2c       = (_Float16*)(ws + WS_W2C);
  _Float16* hbuf      = (_Float16*)(ws + WS_HBUF);
  float*    wsout     = (float*)(ws + WS_OUT);

  moe_convert_x<<<2048, 256, 0, stream>>>(x, xh);
  moe_convert_w<<<4096, 256, 0, stream>>>(w1, w1c, HH, FF);
  moe_convert_w<<<4096, 256, 0, stream>>>(w2, w2c, FF, HH);

  moe_route_count  <<<NCHUNK, 256, 0, stream>>>(tope, partial);
  moe_route_scan   <<<1, 32, 0, stream>>>(partial, meta, chunk_off, tiles);
  moe_route_scatter<<<NCHUNK, 256, 0, stream>>>(tope, chunk_off, sorted);

  moe_gemm1<<<dim3(FF / BN, MAXTIL), 256, 0, stream>>>(xh, w1c, b1, sorted,
                                                       tiles, meta, hbuf);
  moe_gemm2<<<dim3(HH / BN, MAXTIL), 256, 0, stream>>>(hbuf, w2c, b2, sorted,
                                                       tiles, meta, expw, wsout);
  moe_combine<<<TT, 256, 0, stream>>>(wsout, y);
}
